// GraphAttentionTransformer_57294863729159
// MI455X (gfx1250) — matrix operand order repaired and checked by reading, not compiled, not run on hardware
//
#include <hip/hip_runtime.h>

typedef __attribute__((ext_vector_type(16))) _Float16 v16h;
typedef __attribute__((ext_vector_type(8)))  float    v8f;

union F16x16 { v16h v; _Float16 h[16]; };
union F32x8  { v8f  v; float    f[8];  };

__device__ __forceinline__ v8f wmma_f16(v16h a, v16h b, v8f c) {
  // 8 args: (neg_a, A, neg_b, B, c_mod, C, reuse_a, reuse_b)
  return __builtin_amdgcn_wmma_f32_16x16x32_f16(false, a, false, b, (short)0, c, false, false);
}

// hardware v_rcp_f32 / v_rsq_f32 (avoid IEEE div expansion)
__device__ __forceinline__ float frcp(float x) { return __builtin_amdgcn_rcpf(x); }
__device__ __forceinline__ float frsq(float x) { return __builtin_amdgcn_rsqf(x); }

__device__ __forceinline__ float sigm_f(float x) { return frcp(1.f + __expf(-x)); }
__device__ __forceinline__ float silu_f(float x) { return x * sigm_f(x); }

// monotone float <-> uint encoding for atomicMax-based segment max
__device__ __forceinline__ unsigned fenc(float x) {
  unsigned b = __float_as_uint(x);
  return (b & 0x80000000u) ? ~b : (b | 0x80000000u);
}
__device__ __forceinline__ float fdec(unsigned k) {
  unsigned b = (k & 0x80000000u) ? (k & 0x7FFFFFFFu) : ~k;
  return __uint_as_float(b);
}

// C/D fragment: element v -> row = v + 8*(lane>=16), col = lane&15
__device__ __forceinline__ int dRow(int v) {
  return v + (((threadIdx.x & 31) >> 4) << 3);
}

// A fragment (16x32 f16, ISA 7.12.2): lane L: m=L&15; k(j) = base + j + (j>=8 ? 8 : 0),
// base = 32*kt + 8*(L>=16)
__device__ __forceinline__ v16h loadA_h(const _Float16* __restrict__ tile, int stride, int kt) {
  int lane = threadIdx.x & 31;
  const _Float16* p = tile + (size_t)(lane & 15) * stride + 32 * kt + ((lane >> 4) << 3);
  F16x16 a;
  #pragma unroll
  for (int j = 0; j < 8; ++j) { a.h[j] = p[j]; a.h[j + 8] = p[j + 16]; }
  return a.v;
}

__device__ __forceinline__ v16h loadA_f(const float* __restrict__ tile, int stride, int kt, float scale) {
  int lane = threadIdx.x & 31;
  const float* p = tile + (size_t)(lane & 15) * stride + 32 * kt + ((lane >> 4) << 3);
  F16x16 a;
  #pragma unroll
  for (int j = 0; j < 8; ++j) {
    a.h[j]     = (_Float16)(p[j] * scale);
    a.h[j + 8] = (_Float16)(p[j + 16] * scale);
  }
  return a.v;
}

// strided-element f32 A load (for (n,c,i) layouts, es = element stride)
__device__ __forceinline__ v16h loadA_fs(const float* __restrict__ tile, int stride, int es, int kt) {
  int lane = threadIdx.x & 31;
  const float* p = tile + (size_t)(lane & 15) * stride + (size_t)(32 * kt + ((lane >> 4) << 3)) * es;
  F16x16 a;
  #pragma unroll
  for (int j = 0; j < 8; ++j) {
    a.h[j]     = (_Float16)p[j * es];
    a.h[j + 8] = (_Float16)p[(j + 16) * es];
  }
  return a.v;
}

// B fragment from pre-swizzled weights: one contiguous 32B load per lane
__device__ __forceinline__ v16h loadB(const _Float16* __restrict__ swz, int numNt, int kt, int nt) {
  int lane = threadIdx.x & 31;
  return *(const v16h*)(swz + ((size_t)((kt * numNt + nt) * 32 + lane) << 4));
}

// ---------------- kernel 0: weight f32 -> f16 B-fragment swizzle ----------------
__global__ void k_swz(const float* __restrict__ W, _Float16* __restrict__ out,
                      int K, int NC, int srcK) {
  int t = blockIdx.x * 256 + threadIdx.x;
  if (t >= K * NC) return;
  int j = t & 15;
  int lane = (t >> 4) & 31;
  int rem = t >> 9;
  int numNt = NC >> 4;
  int nt = rem % numNt;
  int kt = rem / numNt;
  int k = 32 * kt + ((lane & 16) ? 8 : 0) + (j & 7) + ((j & 8) ? 16 : 0);
  int n = 16 * nt + (lane & 15);
  float v = (k < srcK) ? W[(size_t)k * NC + n] : 0.f;
  out[t] = (_Float16)v;
}

// ---------------- kernel 1: init node state ----------------
__global__ void k_init(const float* __restrict__ embed, const int* __restrict__ atom,
                       float* __restrict__ f0, float* __restrict__ f1,
                       float* __restrict__ agg0, float* __restrict__ agg1,
                       unsigned* __restrict__ amaxu, float* __restrict__ zg, int N) {
  long long t = (long long)blockIdx.x * 256 + threadIdx.x;
  if (t < (long long)N * 192) { f1[t] = 0.f; agg1[t] = 0.f; }
  if (t < (long long)N * 64) {
    agg0[t] = 0.f;
    int n = (int)(t >> 6), c = (int)(t & 63);
    f0[t] = embed[(size_t)atom[n] * 64 + c];
  }
  if (t < (long long)N * 8) { amaxu[t] = 0u; zg[t] = 0.f; }
}

// ---------------- kernel 2: edge geometry + radial MLP + degree scatter ----------------
__global__ __launch_bounds__(32) void k_edge_deg(
    const float* __restrict__ pos, const int* __restrict__ esrc, const int* __restrict__ edst,
    const float* __restrict__ rad_b1, const float* __restrict__ rad_b2,
    const float* __restrict__ rad_bdeg, const float* __restrict__ exp_w,
    const _Float16* __restrict__ zW1, const _Float16* __restrict__ zW2,
    const _Float16* __restrict__ zWdeg, const _Float16* __restrict__ zDp0,
    const _Float16* __restrict__ zDp1,
    float* __restrict__ Y1g, _Float16* __restrict__ hmidg,
    float* __restrict__ f0, float* __restrict__ f1) {
  __shared__ _Float16 sRbf[32][64];
  __shared__ _Float16 sH0[32][64];
  __shared__ _Float16 sH1[32][64];
  __shared__ _Float16 sE0[32][64];
  __shared__ float    sE1[32][64];
  __shared__ float    sY[32][3];
  __shared__ int      sDst[32];

  const int lane = threadIdx.x;
  const long long ebase = (long long)blockIdx.x * 32;
  const long long e = ebase + lane;

  // ---- phase A: per-thread geometry + RBF ----
  {
    int s = esrc[e], d = edst[e];
    sDst[lane] = d;
    float vx = pos[(size_t)d * 3 + 0] - pos[(size_t)s * 3 + 0];
    float vy = pos[(size_t)d * 3 + 1] - pos[(size_t)s * 3 + 1];
    float vz = pos[(size_t)d * 3 + 2] - pos[(size_t)s * 3 + 2];
    float d2 = vx * vx + vy * vy + vz * vz + 1e-9f;
    float id = frsq(d2);            // 1/sqrt(d2), v_rsq_f32
    float dist = d2 * id;           // sqrt(d2)
    float y0 = vx * id, y1 = vy * id, y2 = vz * id;
    Y1g[e * 3 + 0] = y0; Y1g[e * 3 + 1] = y1; Y1g[e * 3 + 2] = y2;
    sY[lane][0] = y0; sY[lane][1] = y1; sY[lane][2] = y2;
    const float start = 0.00673794699f;           // exp(-5)
    const float beta = 633.5086f;                  // (0.04*(1-start))^-2
    float cut = (dist < 5.0f) ? 0.5f * (__cosf(0.6283185307f * dist) + 1.f) : 0.f;
    float ex = __expf(-dist);                      // alpha_s = 1
    for (int i = 0; i < 50; ++i) {
      float mean = start + (1.f - start) * (i * (1.f / 49.f));
      float dl = ex - mean;
      sRbf[lane][i] = (_Float16)(cut * __expf(-beta * dl * dl));
    }
    for (int i = 50; i < 64; ++i) sRbf[lane][i] = (_Float16)0.f;
  }
  __syncthreads();

  const int col0 = lane & 15;
  const float inv_deg = 0.25f;                     // 1/sqrt(16)

  for (int t = 0; t < 2; ++t) {
    const int r0 = 16 * t;
    // GEMM1: rbf @ W1 -> silu -> sH0
    {
      v16h a0 = loadA_h(&sRbf[r0][0], 64, 0);
      v16h a1 = loadA_h(&sRbf[r0][0], 64, 1);
      #pragma unroll
      for (int nt = 0; nt < 4; ++nt) {
        v8f c = {};
        c = wmma_f16(a0, loadB(zW1, 4, 0, nt), c);
        c = wmma_f16(a1, loadB(zW1, 4, 1, nt), c);
        int col = nt * 16 + col0;
        float b = rad_b1[col];
        F32x8 dd; dd.v = c;
        #pragma unroll
        for (int v = 0; v < 8; ++v)
          sH0[r0 + dRow(v)][col] = (_Float16)silu_f(dd.f[v] + b);
      }
    }
    __syncthreads();
    // GEMM2: sH0 @ W2 -> silu -> sH1 (= hmid, also to global f16)
    {
      v16h a0 = loadA_h(&sH0[r0][0], 64, 0);
      v16h a1 = loadA_h(&sH0[r0][0], 64, 1);
      #pragma unroll
      for (int nt = 0; nt < 4; ++nt) {
        v8f c = {};
        c = wmma_f16(a0, loadB(zW2, 4, 0, nt), c);
        c = wmma_f16(a1, loadB(zW2, 4, 1, nt), c);
        int col = nt * 16 + col0;
        float b = rad_b2[col];
        F32x8 dd; dd.v = c;
        #pragma unroll
        for (int v = 0; v < 8; ++v) {
          float h = silu_f(dd.f[v] + b);
          int m = dRow(v);
          sH1[r0 + m][col] = (_Float16)h;
          hmidg[(size_t)(ebase + r0 + m) * 64 + col] = (_Float16)h;
        }
      }
    }
    __syncthreads();
    // GEMM3: hmid @ Wdeg -> e0 (f16) / wd1*exp_w (f32)
    {
      v16h a0 = loadA_h(&sH1[r0][0], 64, 0);
      v16h a1 = loadA_h(&sH1[r0][0], 64, 1);
      #pragma unroll
      for (int nt = 0; nt < 8; ++nt) {
        v8f c = {};
        c = wmma_f16(a0, loadB(zWdeg, 8, 0, nt), c);
        c = wmma_f16(a1, loadB(zWdeg, 8, 1, nt), c);
        int col = nt * 16 + col0;
        float b = rad_bdeg[col];
        F32x8 dd; dd.v = c;
        if (col < 64) {
          float w = exp_w[col];
          #pragma unroll
          for (int v = 0; v < 8; ++v)
            sE0[r0 + dRow(v)][col] = (_Float16)((dd.f[v] + b) * w);
        } else {
          int cc = col - 64;
          float w = exp_w[cc];
          #pragma unroll
          for (int v = 0; v < 8; ++v)
            sE1[r0 + dRow(v)][cc] = (dd.f[v] + b) * w;
        }
      }
    }
    __syncthreads();
    // GEMM4: e0 @ deg_proj0 -> atomic f0
    {
      v16h a0 = loadA_h(&sE0[r0][0], 64, 0);
      v16h a1 = loadA_h(&sE0[r0][0], 64, 1);
      #pragma unroll
      for (int nt = 0; nt < 4; ++nt) {
        v8f c = {};
        c = wmma_f16(a0, loadB(zDp0, 4, 0, nt), c);
        c = wmma_f16(a1, loadB(zDp0, 4, 1, nt), c);
        int col = nt * 16 + col0;
        F32x8 dd; dd.v = c;
        #pragma unroll
        for (int v = 0; v < 8; ++v) {
          int m = dRow(v);
          atomicAdd(&f0[(size_t)sDst[r0 + m] * 64 + col], inv_deg * dd.f[v]);
        }
      }
    }
    // GEMM5: (wd1*exp_w*Y1_i) @ deg_proj1 -> atomic f1 (3 components)
    for (int i = 0; i < 3; ++i) {
      float yi = sY[r0 + (lane & 15)][i];
      v16h a0 = loadA_f(&sE1[r0][0], 64, 0, yi);
      v16h a1 = loadA_f(&sE1[r0][0], 64, 1, yi);
      #pragma unroll
      for (int nt = 0; nt < 4; ++nt) {
        v8f c = {};
        c = wmma_f16(a0, loadB(zDp1, 4, 0, nt), c);
        c = wmma_f16(a1, loadB(zDp1, 4, 1, nt), c);
        int col = nt * 16 + col0;
        F32x8 dd; dd.v = c;
        #pragma unroll
        for (int v = 0; v < 8; ++v) {
          int m = dRow(v);
          atomicAdd(&f1[((size_t)sDst[r0 + m] * 64 + col) * 3 + i], inv_deg * dd.f[v]);
        }
      }
    }
    __syncthreads();
  }
}

// ---------------- kernels 3/5: w_attn (WMMA) + messages ----------------
template <bool AGG>
__global__ __launch_bounds__(32) void k_attn(
    const int* __restrict__ esrc, const int* __restrict__ edst,
    const _Float16* __restrict__ hmidg, const _Float16* __restrict__ zWattn,
    const float* __restrict__ battn, const float* __restrict__ alpha_dot,
    const float* __restrict__ Y1g, const float* __restrict__ f0g, const float* __restrict__ f1g,
    float* __restrict__ ag, unsigned* __restrict__ amaxu,
    const float* __restrict__ eag, const float* __restrict__ zg,
    float* __restrict__ agg0, float* __restrict__ agg1) {
  __shared__ float sWA[32][320];
  const int lane = threadIdx.x;
  const long long ebase = (long long)blockIdx.x * 32;
  const long long e = ebase + lane;
  const int col0 = lane & 15;

  // w_attn = hmid @ Wattn + battn  (2 M-tiles x 20 N-tiles)
  for (int t = 0; t < 2; ++t) {
    const _Float16* hp = hmidg + (size_t)(ebase + 16 * t) * 64;
    v16h a0 = loadA_h(hp, 64, 0);
    v16h a1 = loadA_h(hp, 64, 1);
    #pragma unroll
    for (int nt = 0; nt < 20; ++nt) {
      v8f c = {};
      c = wmma_f16(a0, loadB(zWattn, 20, 0, nt), c);
      c = wmma_f16(a1, loadB(zWattn, 20, 1, nt), c);
      int col = nt * 16 + col0;
      float b = battn[col];
      F32x8 dd; dd.v = c;
      #pragma unroll
      for (int v = 0; v < 8; ++v) sWA[16 * t + dRow(v)][col] = dd.f[v] + b;
    }
  }
  __syncthreads();

  const int s = esrc[e], d = edst[e];
  const float y0 = Y1g[e * 3 + 0], y1 = Y1g[e * 3 + 1], y2 = Y1g[e * 3 + 2];
  const float is3 = 0.5773502692f, is2 = 0.7071067812f;

  if (!AGG) {
    float acc[8] = {0.f, 0.f, 0.f, 0.f, 0.f, 0.f, 0.f, 0.f};
    for (int c = 0; c < 64; ++c) {
      float s0c = f0g[(size_t)s * 64 + c];
      const float* s1p = f1g + ((size_t)s * 64 + c) * 3;
      float dotv = s1p[0] * y0 + s1p[1] * y1 + s1p[2] * y2;
      float m0 = sWA[lane][c] * s0c + sWA[lane][192 + c] * dotv * is3;
      float slr = 0.2f * m0 + 0.8f * m0 * sigm_f(m0);
      acc[c >> 3] += slr * alpha_dot[(c >> 3) * 8 + (c & 7)];
    }
    #pragma unroll
    for (int h = 0; h < 8; ++h) {
      ag[e * 8 + h] = acc[h];
      atomicMax(&amaxu[(size_t)d * 8 + h], fenc(acc[h]));
    }
  } else {
    float attn[8];
    #pragma unroll
    for (int h = 0; h < 8; ++h) attn[h] = eag[e * 8 + h] * frcp(zg[(size_t)d * 8 + h]);
    for (int c = 0; c < 64; ++c) {
      float s0c = f0g[(size_t)s * 64 + c];
      const float* s1p = f1g + ((size_t)s * 64 + c) * 3;
      float s1x = s1p[0], s1y = s1p[1], s1z = s1p[2];
      float dotv = s1x * y0 + s1y * y1 + s1z * y2;
      float m0 = sWA[lane][c] * s0c + sWA[lane][192 + c] * dotv * is3;
      float w1 = sWA[lane][64 + c], w2 = sWA[lane][128 + c], w4 = sWA[lane][256 + c] * is2;
      float cx = s1y * y2 - s1z * y1;
      float cy = s1z * y0 - s1x * y2;
      float cz = s1x * y1 - s1y * y0;
      float b1 = w1 * s0c;
      float m1x = b1 * y0 + w2 * s1x + w4 * cx;
      float m1y = b1 * y1 + w2 * s1y + w4 * cy;
      float m1z = b1 * y2 + w2 * s1z + w4 * cz;
      float at = attn[c >> 3];
      atomicAdd(&agg0[(size_t)d * 64 + c], at * m0);
      float* p = &agg1[((size_t)d * 64 + c) * 3];
      atomicAdd(p + 0, at * m1x);
      atomicAdd(p + 1, at * m1y);
      atomicAdd(p + 2, at * m1z);
    }
  }
}

// ---------------- kernel 4: softmax denominator ----------------
__global__ void k_softz(const int* __restrict__ edst, const float* __restrict__ ag,
                        const unsigned* __restrict__ amaxu,
                        float* __restrict__ eag, float* __restrict__ zg, long long E) {
  long long t = (long long)blockIdx.x * 256 + threadIdx.x;
  if (t >= E) return;
  int d = edst[t];
  #pragma unroll
  for (int h = 0; h < 8; ++h) {
    float mx = fdec(amaxu[(size_t)d * 8 + h]);
    float ev = __expf(ag[t * 8 + h] - mx);
    eag[t * 8 + h] = ev;
    atomicAdd(&zg[(size_t)d * 8 + h], ev);
  }
}

// ---------------- kernel 6: node update (out_proj + FFN) ----------------
__global__ __launch_bounds__(32) void k_node(
    const float* __restrict__ f0g, const float* __restrict__ f1g,
    const float* __restrict__ agg0, const float* __restrict__ agg1,
    const _Float16* __restrict__ zOp0, const _Float16* __restrict__ zOp1,
    const _Float16* __restrict__ zF1s, const _Float16* __restrict__ zF1v,
    const _Float16* __restrict__ zF2s, const _Float16* __restrict__ zF2v,
    const float* __restrict__ b1s, const float* __restrict__ b2s,
    float* __restrict__ out) {
  __shared__ float    sF0[16][64];
  __shared__ float    sF1[3][16][64];
  __shared__ _Float16 sAh[16][64];
  __shared__ float    sGate[16][64];

  const int lane = threadIdx.x;
  const size_t n0 = (size_t)blockIdx.x * 16;
  const int col0 = lane & 15;

  // stage 1: f0 += agg0 @ out_proj0
  {
    const float* ap = agg0 + n0 * 64;
    v16h a0 = loadA_f(ap, 64, 0, 1.f);
    v16h a1 = loadA_f(ap, 64, 1, 1.f);
    #pragma unroll
    for (int nt = 0; nt < 4; ++nt) {
      int col = nt * 16 + col0;
      F32x8 c;
      #pragma unroll
      for (int v = 0; v < 8; ++v) c.f[v] = f0g[(n0 + dRow(v)) * 64 + col];
      v8f dacc = wmma_f16(a0, loadB(zOp0, 4, 0, nt), c.v);
      dacc = wmma_f16(a1, loadB(zOp0, 4, 1, nt), dacc);
      F32x8 dd; dd.v = dacc;
      #pragma unroll
      for (int v = 0; v < 8; ++v) sF0[dRow(v)][col] = dd.f[v];
    }
  }
  // stage 2: f1 += agg1 @ out_proj1 (per component)
  for (int i = 0; i < 3; ++i) {
    const float* ap = agg1 + n0 * 192 + i;
    v16h a0 = loadA_fs(ap, 192, 3, 0);
    v16h a1 = loadA_fs(ap, 192, 3, 1);
    #pragma unroll
    for (int nt = 0; nt < 4; ++nt) {
      int col = nt * 16 + col0;
      F32x8 c;
      #pragma unroll
      for (int v = 0; v < 8; ++v) c.f[v] = f1g[(n0 + dRow(v)) * 192 + (size_t)col * 3 + i];
      v8f dacc = wmma_f16(a0, loadB(zOp1, 4, 0, nt), c.v);
      dacc = wmma_f16(a1, loadB(zOp1, 4, 1, nt), dacc);
      F32x8 dd; dd.v = dacc;
      #pragma unroll
      for (int v = 0; v < 8; ++v) sF1[i][dRow(v)][col] = dd.f[v];
    }
  }
  __syncthreads();
  // stage 3: hs = f0 @ ffn_W1s + b1s -> mid_s (silu) | gate (sigmoid)
  {
    v16h a0 = loadA_f(&sF0[0][0], 64, 0, 1.f);
    v16h a1 = loadA_f(&sF0[0][0], 64, 1, 1.f);
    #pragma unroll
    for (int nt = 0; nt < 8; ++nt) {
      int col = nt * 16 + col0;
      F32x8 c;
      float b = b1s[col];
      #pragma unroll
      for (int v = 0; v < 8; ++v) c.f[v] = b;
      v8f dacc = wmma_f16(a0, loadB(zF1s, 8, 0, nt), c.v);
      dacc = wmma_f16(a1, loadB(zF1s, 8, 1, nt), dacc);
      F32x8 dd; dd.v = dacc;
      if (nt < 4) {
        #pragma unroll
        for (int v = 0; v < 8; ++v) sAh[dRow(v)][col] = (_Float16)silu_f(dd.f[v]);
      } else {
        #pragma unroll
        for (int v = 0; v < 8; ++v) sGate[dRow(v)][col - 64] = sigm_f(dd.f[v]);
      }
    }
  }
  __syncthreads();
  // stage 4: f0_out = f0 + mid_s @ ffn_W2s + b2s
  {
    v16h a0 = loadA_h(&sAh[0][0], 64, 0);
    v16h a1 = loadA_h(&sAh[0][0], 64, 1);
    #pragma unroll
    for (int nt = 0; nt < 4; ++nt) {
      int col = nt * 16 + col0;
      F32x8 c;
      float b = b2s[col];
      #pragma unroll
      for (int v = 0; v < 8; ++v) c.f[v] = sF0[dRow(v)][col] + b;
      v8f dacc = wmma_f16(a0, loadB(zF2s, 4, 0, nt), c.v);
      dacc = wmma_f16(a1, loadB(zF2s, 4, 1, nt), dacc);
      F32x8 dd; dd.v = dacc;
      #pragma unroll
      for (int v = 0; v < 8; ++v) out[(n0 + dRow(v)) * 256 + col] = dd.f[v];
    }
  }
  __syncthreads();
  // stage 5: mid_v = (f1 @ ffn_W1v)*gate ; f1_out = f1 + mid_v @ ffn_W2v
  for (int i = 0; i < 3; ++i) {
    {
      v16h a0 = loadA_f(&sF1[i][0][0], 64, 0, 1.f);
      v16h a1 = loadA_f(&sF1[i][0][0], 64, 1, 1.f);
      #pragma unroll
      for (int nt = 0; nt < 4; ++nt) {
        int col = nt * 16 + col0;
        v8f dacc = {};
        dacc = wmma_f16(a0, loadB(zF1v, 4, 0, nt), dacc);
        dacc = wmma_f16(a1, loadB(zF1v, 4, 1, nt), dacc);
        F32x8 dd; dd.v = dacc;
        #pragma unroll
        for (int v = 0; v < 8; ++v) {
          int m = dRow(v);
          sAh[m][col] = (_Float16)(dd.f[v] * sGate[m][col]);
        }
      }
    }
    __syncthreads();
    {
      v16h a0 = loadA_h(&sAh[0][0], 64, 0);
      v16h a1 = loadA_h(&sAh[0][0], 64, 1);
      #pragma unroll
      for (int nt = 0; nt < 4; ++nt) {
        int col = nt * 16 + col0;
        F32x8 c;
        #pragma unroll
        for (int v = 0; v < 8; ++v) c.f[v] = sF1[i][dRow(v)][col];
        v8f dacc = wmma_f16(a0, loadB(zF2v, 4, 0, nt), c.v);
        dacc = wmma_f16(a1, loadB(zF2v, 4, 1, nt), dacc);
        F32x8 dd; dd.v = dacc;
        #pragma unroll
        for (int v = 0; v < 8; ++v)
          out[(n0 + dRow(v)) * 256 + 64 + (size_t)col * 3 + i] = dd.f[v];
      }
    }
    __syncthreads();
  }
}

// ---------------- host launcher ----------------
extern "C" void kernel_launch(void* const* d_in, const int* in_sizes, int n_in,
                              void* d_out, int out_size, void* d_ws, size_t ws_size,
                              hipStream_t stream) {
  const float* pos       = (const float*)d_in[0];
  const float* embed_W   = (const float*)d_in[1];
  const float* exp_w     = (const float*)d_in[2];
  const float* rad_W1    = (const float*)d_in[3];
  const float* rad_b1    = (const float*)d_in[4];
  const float* rad_W2    = (const float*)d_in[5];
  const float* rad_b2    = (const float*)d_in[6];
  const float* rad_Wdeg  = (const float*)d_in[7];
  const float* rad_bdeg  = (const float*)d_in[8];
  const float* rad_Wattn = (const float*)d_in[9];
  const float* rad_battn = (const float*)d_in[10];
  const float* deg_proj0 = (const float*)d_in[11];
  const float* deg_proj1 = (const float*)d_in[12];
  const float* alpha_dot = (const float*)d_in[13];
  const float* out_proj0 = (const float*)d_in[14];
  const float* out_proj1 = (const float*)d_in[15];
  const float* ffn_W1s   = (const float*)d_in[16];
  const float* ffn_b1s   = (const float*)d_in[17];
  const float* ffn_W1v   = (const float*)d_in[18];
  const float* ffn_W2s   = (const float*)d_in[19];
  const float* ffn_b2s   = (const float*)d_in[20];
  const float* ffn_W2v   = (const float*)d_in[21];
  const int* node_atom   = (const int*)d_in[22];
  const int* edge_src    = (const int*)d_in[23];
  const int* edge_dst    = (const int*)d_in[24];
  (void)n_in; (void)out_size; (void)ws_size;

  const int N = in_sizes[22];
  const int E = in_sizes[23];

  char* ws = (char*)d_ws;
  size_t off = 0;
  auto take = [&](size_t bytes) -> void* {
    void* p = (void*)(ws + off);
    off += (bytes + 255) & ~(size_t)255;
    return p;
  };
  _Float16* swz  = (_Float16*)take((size_t)73728 * 2);
  float*    f0   = (float*)take((size_t)N * 64 * 4);
  float*    f1   = (float*)take((size_t)N * 192 * 4);
  float*    Y1   = (float*)take((size_t)E * 3 * 4);
  _Float16* hmid = (_Float16*)take((size_t)E * 64 * 2);
  float*    ag   = (float*)take((size_t)E * 8 * 4);
  float*    eag  = (float*)take((size_t)E * 8 * 4);
  unsigned* amaxu = (unsigned*)take((size_t)N * 8 * 4);
  float*    zg   = (float*)take((size_t)N * 8 * 4);
  float*    agg0 = (float*)take((size_t)N * 64 * 4);
  float*    agg1 = (float*)take((size_t)N * 192 * 4);

  _Float16* zW1    = swz + 0;
  _Float16* zW2    = swz + 4096;
  _Float16* zWdeg  = swz + 8192;
  _Float16* zWattn = swz + 16384;
  _Float16* zDp0   = swz + 36864;
  _Float16* zDp1   = swz + 40960;
  _Float16* zOp0   = swz + 45056;
  _Float16* zOp1   = swz + 49152;
  _Float16* zF1s   = swz + 53248;
  _Float16* zF1v   = swz + 61440;
  _Float16* zF2s   = swz + 65536;
  _Float16* zF2v   = swz + 69632;

  // weight swizzle (f32 -> f16 B-fragment layout)
  k_swz<<<16, 256, 0, stream>>>(rad_W1,    zW1,    64, 64, 50);
  k_swz<<<16, 256, 0, stream>>>(rad_W2,    zW2,    64, 64, 64);
  k_swz<<<32, 256, 0, stream>>>(rad_Wdeg,  zWdeg,  64, 128, 64);
  k_swz<<<80, 256, 0, stream>>>(rad_Wattn, zWattn, 64, 320, 64);
  k_swz<<<16, 256, 0, stream>>>(deg_proj0, zDp0,   64, 64, 64);
  k_swz<<<16, 256, 0, stream>>>(deg_proj1, zDp1,   64, 64, 64);
  k_swz<<<16, 256, 0, stream>>>(out_proj0, zOp0,   64, 64, 64);
  k_swz<<<16, 256, 0, stream>>>(out_proj1, zOp1,   64, 64, 64);
  k_swz<<<32, 256, 0, stream>>>(ffn_W1s,   zF1s,   64, 128, 64);
  k_swz<<<16, 256, 0, stream>>>(ffn_W1v,   zF1v,   64, 64, 64);
  k_swz<<<16, 256, 0, stream>>>(ffn_W2s,   zF2s,   64, 64, 64);
  k_swz<<<16, 256, 0, stream>>>(ffn_W2v,   zF2v,   64, 64, 64);

  k_init<<<(int)(((long long)N * 192 + 255) / 256), 256, 0, stream>>>(
      embed_W, node_atom, f0, f1, agg0, agg1, amaxu, zg, N);

  k_edge_deg<<<E / 32, 32, 0, stream>>>(
      pos, edge_src, edge_dst, rad_b1, rad_b2, rad_bdeg, exp_w,
      zW1, zW2, zWdeg, zDp0, zDp1, Y1, hmid, f0, f1);

  k_attn<false><<<E / 32, 32, 0, stream>>>(
      edge_src, edge_dst, hmid, zWattn, rad_battn, alpha_dot, Y1, f0, f1,
      ag, amaxu, eag, zg, agg0, agg1);

  k_softz<<<(E + 255) / 256, 256, 0, stream>>>(edge_dst, ag, amaxu, eag, zg, (long long)E);

  k_attn<true><<<E / 32, 32, 0, stream>>>(
      edge_src, edge_dst, hmid, zWattn, rad_battn, alpha_dot, Y1, f0, f1,
      ag, amaxu, eag, zg, agg0, agg1);

  k_node<<<N / 16, 32, 0, stream>>>(
      f0, f1, agg0, agg1, zOp0, zOp1, zF1s, zF1v, zF2s, zF2v,
      ffn_b1s, ffn_b2s, (float*)d_out);
}
